// RandomAttention_3418793968317
// MI455X (gfx1250) — matrix operand order repaired and checked by reading, not compiled, not run on hardware
//
#include <hip/hip_runtime.h>
#include <hip/hip_bf16.h>

typedef __attribute__((ext_vector_type(16))) __bf16 v16bf;
typedef __attribute__((ext_vector_type(8)))  __bf16 v8bf;
typedef __attribute__((ext_vector_type(4)))  __bf16 v4bf;
typedef __attribute__((ext_vector_type(8)))  float  v8f;

#define L_DIM   1024
#define DK_DIM  128
#define H_DIM   16
#define B_DIM   4
#define BH_DIM  (B_DIM * H_DIM)
#define FSTRIDE 1028   // f32 tile row stride (16B aligned, 4-bank skew/row)
#define BSTRIDE 1032   // bf16 tile row stride (16B aligned, 4-bank skew/row)
#define NEG_BIG -1.0e9f
#define KEEP_SCALE (1.0f / 0.9f)
#define DROP_THRESH 1677722u   // 0.1 * 2^24; drop if hash24 < this
#define VT_BYTES ((size_t)BH_DIM * DK_DIM * L_DIM * 2)

// ---------------- kernel 1: V (bh,k,d) f32  ->  Vt (bh,d,k) bf16 ----------------
__global__ __launch_bounds__(256) void v_to_bf16_t(const float* __restrict__ v,
                                                   __bf16* __restrict__ vt) {
    __shared__ __bf16 t[32][33];
    const int bh = blockIdx.z;
    const int d0 = blockIdx.x * 32;
    const int k0 = blockIdx.y * 32;
    const int tx = threadIdx.x, ty = threadIdx.y;   // 32 x 8
    const float* src = v + (size_t)bh * L_DIM * DK_DIM;
    #pragma unroll
    for (int j = 0; j < 32; j += 8)
        t[ty + j][tx] = (__bf16)src[(size_t)(k0 + ty + j) * DK_DIM + d0 + tx];
    __syncthreads();
    __bf16* dst = vt + (size_t)bh * DK_DIM * L_DIM;
    #pragma unroll
    for (int j = 0; j < 32; j += 8)
        dst[(size_t)(d0 + ty + j) * L_DIM + k0 + tx] = t[tx][ty + j];
}

// ---------------- kernel 2: fused masked-softmax + dropout + P@V ----------------
__global__ __launch_bounds__(128) void rand_attn_fused(
    const float*  __restrict__ vmat_g,   // (B,H,L,DK) f32
    const __bf16* __restrict__ vt_g,     // (B,H,DK,L) bf16 (may be unused)
    const int*    __restrict__ mask_g,   // (B,1,L,L)  i32
    const float*  __restrict__ rattn_g,  // (B,H,L,L)  f32
    float*        __restrict__ out_g,    // (B,H,L,DK) f32
    float*        __restrict__ attn_g,   // (B,H,L,L)  f32
    int use_vt)
{
    __shared__ float  sPf[16 * FSTRIDE];   // masked logits -> probs (f32)
    __shared__ __bf16 sPb[16 * BSTRIDE];   // post-dropout probs (bf16) for WMMA A

    const int tile = blockIdx.x;                 // B*H*(L/16) tiles
    const int qt   = tile & ((L_DIM / 16) - 1);
    const int bh   = tile >> 6;                  // L/16 == 64
    const int b    = bh / H_DIM;
    const int q0   = qt * 16;

    const int tid   = threadIdx.x;   // 0..127
    const int row   = tid >> 3;      // 0..15 : q-row
    const int lane8 = tid & 7;       // 8 lanes per row

    const float4* arow4 = (const float4*)(rattn_g + ((size_t)bh * L_DIM + q0 + row) * L_DIM);
    const int4*   mrow4 = (const int4*)(mask_g  + ((size_t)b  * L_DIM + q0 + row) * L_DIM);
    float*  sfr = sPf + row * FSTRIDE;
    __bf16* sbr = sPb + row * BSTRIDE;

    // ---- pass 1: masked load -> LDS (f32), row max; float4 per iteration ----
    float mx = -3.0e38f;
    for (int i = lane8; i < L_DIM / 4; i += 8) {
        float4 x = arow4[i];
        int4   m = mrow4[i];
        x.x = (m.x == 0) ? NEG_BIG : x.x;
        x.y = (m.y == 0) ? NEG_BIG : x.y;
        x.z = (m.z == 0) ? NEG_BIG : x.z;
        x.w = (m.w == 0) ? NEG_BIG : x.w;
        *(float4*)(sfr + i * 4) = x;
        mx = fmaxf(mx, fmaxf(fmaxf(x.x, x.y), fmaxf(x.z, x.w)));
    }
    #pragma unroll
    for (int off = 1; off < 8; off <<= 1)
        mx = fmaxf(mx, __shfl_xor(mx, off, 32));

    // ---- pass 2: exp + row sum ----
    float sum = 0.0f;
    for (int i = lane8; i < L_DIM / 4; i += 8) {
        float4 x = *(float4*)(sfr + i * 4);
        x.x = __expf(x.x - mx);
        x.y = __expf(x.y - mx);
        x.z = __expf(x.z - mx);
        x.w = __expf(x.w - mx);
        *(float4*)(sfr + i * 4) = x;
        sum += (x.x + x.y) + (x.z + x.w);
    }
    #pragma unroll
    for (int off = 1; off < 8; off <<= 1)
        sum += __shfl_xor(sum, off, 32);
    const float inv = 1.0f / sum;

    // ---- pass 3: normalize + dropout; write attn f32 (global) + bf16 (LDS) ----
    float4* aout4 = (float4*)(attn_g + ((size_t)bh * L_DIM + q0 + row) * L_DIM);
    const unsigned base = ((unsigned)(bh * L_DIM + q0 + row)) * (unsigned)L_DIM;
    for (int i = lane8; i < L_DIM / 4; i += 8) {
        float4 p = *(float4*)(sfr + i * 4);
        p.x *= inv; p.y *= inv; p.z *= inv; p.w *= inv;
        const unsigned k = (unsigned)(i * 4);
        #pragma unroll
        for (int e = 0; e < 4; ++e) {
            unsigned h = (base + k + e) * 2654435761u;
            h ^= h >> 13;
            float* pe = (&p.x) + e;
            *pe = ((h & 0xFFFFFFu) >= DROP_THRESH) ? *pe * KEEP_SCALE : 0.0f;
        }
        aout4[i] = p;
        v4bf pb = { (__bf16)p.x, (__bf16)p.y, (__bf16)p.z, (__bf16)p.w };
        *(v4bf*)(sbr + i * 4) = pb;
    }
    __syncthreads();

    // ---- matmul: out[16xDK] = P[16xL] @ V[LxDK] via bf16 WMMA ----
    const int wave  = tid >> 5;      // 0..3 : owns DK cols [wave*32, wave*32+32)
    const int lane  = tid & 31;
    const int lhalf = lane >> 4;     // 0 or 1
    const int l16   = lane & 15;
    const int nb    = wave * 32;

    v8f acc0 = {};
    v8f acc1 = {};
    const __bf16* abase = sPb + l16 * BSTRIDE + lhalf * 16;

    if (use_vt) {
        // B rows contiguous in Vt: lane l16 -> column d = nb+l16, 16 consecutive k
        const __bf16* bp0 = vt_g + ((size_t)bh * DK_DIM + nb + l16) * L_DIM + lhalf * 16;
        const __bf16* bp1 = bp0 + (size_t)16 * L_DIM;
        for (int k0 = 0; k0 < L_DIM; k0 += 32) {
            v8bf alo = *(const v8bf*)(abase + k0);        // K = k0+lhalf*16 ..+8
            v8bf ahi = *(const v8bf*)(abase + k0 + 8);    // K = k0+lhalf*16+8 ..+8
            v16bf a = __builtin_shufflevector(alo, ahi, 0,1,2,3,4,5,6,7,
                                              8,9,10,11,12,13,14,15);
            v8bf b0lo = *(const v8bf*)(bp0 + k0);
            v8bf b0hi = *(const v8bf*)(bp0 + k0 + 8);
            v8bf b1lo = *(const v8bf*)(bp1 + k0);
            v8bf b1hi = *(const v8bf*)(bp1 + k0 + 8);
            v16bf b0 = __builtin_shufflevector(b0lo, b0hi, 0,1,2,3,4,5,6,7,
                                               8,9,10,11,12,13,14,15);
            v16bf b1 = __builtin_shufflevector(b1lo, b1hi, 0,1,2,3,4,5,6,7,
                                               8,9,10,11,12,13,14,15);
            acc0 = __builtin_amdgcn_wmma_f32_16x16x32_bf16(
                false, a, false, b0, (short)0, acc0, false, false);
            acc1 = __builtin_amdgcn_wmma_f32_16x16x32_bf16(
                false, a, false, b1, (short)0, acc1, false, false);
        }
    } else {
        // fallback: gather f32 V directly and convert
        const float* vmat = vmat_g + (size_t)bh * L_DIM * DK_DIM;
        for (int k0 = 0; k0 < L_DIM; k0 += 32) {
            v8bf alo = *(const v8bf*)(abase + k0);
            v8bf ahi = *(const v8bf*)(abase + k0 + 8);
            v16bf a = __builtin_shufflevector(alo, ahi, 0,1,2,3,4,5,6,7,
                                              8,9,10,11,12,13,14,15);
            v16bf b0, b1;
            const float* vp = vmat + (size_t)(k0 + lhalf * 16) * DK_DIM + nb + l16;
            #pragma unroll
            for (int j = 0; j < 16; ++j) b0[j] = (__bf16)vp[(size_t)j * DK_DIM];
            #pragma unroll
            for (int j = 0; j < 16; ++j) b1[j] = (__bf16)vp[(size_t)j * DK_DIM + 16];
            acc0 = __builtin_amdgcn_wmma_f32_16x16x32_bf16(
                false, a, false, b0, (short)0, acc0, false, false);
            acc1 = __builtin_amdgcn_wmma_f32_16x16x32_bf16(
                false, a, false, b1, (short)0, acc1, false, false);
        }
    }

    // C/D f32 16x16 layout: lane l16 -> N=l16; VGPR r -> M=r (+8 for lanes 16-31)
    float* orow = out_g + ((size_t)bh * L_DIM + q0) * DK_DIM;
    #pragma unroll
    for (int r = 0; r < 8; ++r) {
        const int m = lhalf * 8 + r;
        orow[(size_t)m * DK_DIM + nb + l16]      = acc0[r];
        orow[(size_t)m * DK_DIM + nb + 16 + l16] = acc1[r];
    }
}

extern "C" void kernel_launch(void* const* d_in, const int* in_sizes, int n_in,
                              void* d_out, int out_size, void* d_ws, size_t ws_size,
                              hipStream_t stream) {
    (void)in_sizes; (void)n_in; (void)out_size;
    const float* v     = (const float*)d_in[0];
    // d_in[1] = len_q scalar (fixed 1024 for this shape)
    const int*   mask  = (const int*)d_in[2];
    const float* rattn = (const float*)d_in[3];

    float* out  = (float*)d_out;
    float* attn = out + (size_t)B_DIM * H_DIM * L_DIM * DK_DIM;

    __bf16* vt = (__bf16*)d_ws;
    const int use_vt = (d_ws != nullptr && ws_size >= VT_BYTES) ? 1 : 0;

    if (use_vt) {
        dim3 tgrid(DK_DIM / 32, L_DIM / 32, BH_DIM);   // (4, 32, 64)
        v_to_bf16_t<<<tgrid, dim3(32, 8), 0, stream>>>(v, vt);
    }

    const int nblocks = BH_DIM * (L_DIM / 16);         // 4096
    rand_attn_fused<<<nblocks, 128, 0, stream>>>(v, vt, mask, rattn, out, attn, use_vt);
}
